// UserEncoder_80118319940241
// MI455X (gfx1250) — matrix operand order repaired and checked by reading, not compiled, not run on hardware
//
#include <hip/hip_runtime.h>

#define N_NODES 32768
#define EMBED_D 512
#define NNZ (N_NODES * 32)

typedef __bf16 bf16_t;
typedef bf16_t v16bf __attribute__((ext_vector_type(16)));
typedef float  v8f   __attribute__((ext_vector_type(8)));

union Frag16 {
    v16bf v;
    uint4 q[2];
};

// Native f32->bf16 (RNE) pack; backend emits v_cvt_pk_bf16_f32.
__device__ __forceinline__ unsigned int pack_bf16(float a, float b) {
    union { bf16_t h[2]; unsigned int u; } o;
    o.h[0] = (bf16_t)a;
    o.h[1] = (bf16_t)b;
    return o.u;
}

// ---------------------------------------------------------------- W -> bf16
__global__ __launch_bounds__(256) void convert_w_kernel(
    const float* __restrict__ W, unsigned short* __restrict__ Wbf) {
    int idx = blockIdx.x * 256 + threadIdx.x;        // 65536 threads, 4 elems each
    float4 f = ((const float4*)W)[idx];
    uint2 o;
    o.x = pack_bf16(f.x, f.y);
    o.y = pack_bf16(f.z, f.w);
    ((uint2*)Wbf)[idx] = o;
}

// ---------------------------------------------------------------- zero agg
__global__ __launch_bounds__(256) void zero_kernel(float4* __restrict__ p) {
    long i = (long)blockIdx.x * 256 + threadIdx.x;
    p[i] = make_float4(0.f, 0.f, 0.f, 0.f);
}

// ---------------------------------------------------------------- gather + GEMM (WMMA bf16)
// support[n, :] = embedding[user_ids[n], :] @ W      (M=32768, K=N=512)
// Double-buffered LDS ping-pong: one barrier per K-step; global loads for
// tile k+1 issue before the WMMAs of tile k. Each thread's gathered row is
// fixed, so the row pointer is derived straight from kernarg pointers
// (keeps loads GLOBAL, not FLAT, so they track LOADcnt only).
#define BM 128
#define BN 64
#define BK 32
#define A_STRIDE 40   // ushorts per row (80B, 16B aligned, conflict-spreading pad)
#define B_STRIDE 40

__global__ __launch_bounds__(256) void gemm_gather_kernel(
    const int*            __restrict__ user_ids,
    const float*          __restrict__ embedding,
    const unsigned short* __restrict__ Wbf,      // [K=512][N=512] bf16
    float*                __restrict__ support)  // [32768][512] f32
{
    __shared__ unsigned short Alds[2][BM * A_STRIDE];
    __shared__ unsigned short Blds[2][BN * B_STRIDE];   // transposed: [n][k]

    const int tid = threadIdx.x;
    const int nb  = blockIdx.x * BN;
    const int mb  = blockIdx.y * BM;

    const int wave   = tid >> 5;
    const int lane   = tid & 31;
    const int khalf  = lane >> 4;     // 0 or 1
    const int lane16 = lane & 15;
    const int mw = (wave >> 1) * 32;  // wave's M offset in block tile
    const int nw = (wave & 1)  * 32;  // wave's N offset in block tile

    // staging assignments
    const int ar  = tid >> 1;         // A row (2 threads/row)
    const int akk = (tid & 1) * 16;   // A k-offset within tile
    const int bnl = tid & 63;         // B n_local
    const int bkb = (tid >> 6) * 8;   // B k-base (8 consecutive k)

    // per-thread gathered row base: straight from kernarg pointers -> GLOBAL loads
    const float* myrow = embedding + (long)user_ids[mb + ar] * EMBED_D + akk;

    v8f acc[2][2];
    v8f vzero = {};
    #pragma unroll
    for (int i = 0; i < 2; ++i)
        #pragma unroll
        for (int j = 0; j < 2; ++j) acc[i][j] = vzero;

    // ---- prologue: fetch tile 0 into registers
    float4         af[4];
    unsigned short br[8];
    {
        #pragma unroll
        for (int i = 0; i < 4; ++i) af[i] = ((const float4*)myrow)[i];
        const unsigned short* bsrc = Wbf + (long)bkb * EMBED_D + nb + bnl;
        #pragma unroll
        for (int i = 0; i < 8; ++i) br[i] = bsrc[(long)i * EMBED_D];
    }

    for (int k0 = 0; k0 < EMBED_D; k0 += BK) {
        const int buf = (k0 >> 5) & 1;

        // ---- commit registers -> LDS[buf]
        {
            unsigned int* adst = (unsigned int*)&Alds[buf][ar * A_STRIDE + akk];
            #pragma unroll
            for (int i = 0; i < 4; ++i) {
                adst[i * 2 + 0] = pack_bf16(af[i].x, af[i].y);
                adst[i * 2 + 1] = pack_bf16(af[i].z, af[i].w);
            }
            unsigned short* bdst = &Blds[buf][bnl * B_STRIDE + bkb];
            #pragma unroll
            for (int i = 0; i < 8; ++i) bdst[i] = br[i];
        }
        __syncthreads();

        // ---- issue global loads for tile k0+BK (latency hides under WMMAs)
        const int kn = k0 + BK;
        if (kn < EMBED_D) {
            const float* asrc = myrow + kn;
            #pragma unroll
            for (int i = 0; i < 4; ++i) af[i] = ((const float4*)asrc)[i];
            const unsigned short* bsrc = Wbf + (long)(kn + bkb) * EMBED_D + nb + bnl;
            #pragma unroll
            for (int i = 0; i < 8; ++i) br[i] = bsrc[(long)i * EMBED_D];
        }

        // ---- fragments (A 16x32 & K-major B, both contiguous k = 16*khalf + s)
        Frag16 a[2], b[2];
        #pragma unroll
        for (int i = 0; i < 2; ++i) {
            const unsigned short* rp = &Alds[buf][(mw + i * 16 + lane16) * A_STRIDE];
            a[i].q[0] = *(const uint4*)(rp + khalf * 16);       // K = 16*khalf .. +7
            a[i].q[1] = *(const uint4*)(rp + khalf * 16 + 8);   // K = 16*khalf+8 .. +15
        }
        #pragma unroll
        for (int j = 0; j < 2; ++j) {
            const unsigned short* cp = &Blds[buf][(nw + j * 16 + lane16) * B_STRIDE + khalf * 16];
            b[j].q[0] = ((const uint4*)cp)[0];                  // K = 16*khalf .. +7
            b[j].q[1] = ((const uint4*)cp)[1];                  // .. +15
        }
        #pragma unroll
        for (int i = 0; i < 2; ++i)
            #pragma unroll
            for (int j = 0; j < 2; ++j)
                acc[i][j] = __builtin_amdgcn_wmma_f32_16x16x32_bf16(
                    false, a[i].v, false, b[j].v, (short)0, acc[i][j], false, false);
    }

    // ---- epilogue: C layout -> row-major f32
    #pragma unroll
    for (int i = 0; i < 2; ++i)
        #pragma unroll
        for (int j = 0; j < 2; ++j) {
            int col = nb + nw + j * 16 + lane16;
            #pragma unroll
            for (int r = 0; r < 8; ++r) {
                int row = mb + mw + i * 16 + r + khalf * 8;
                support[(long)row * EMBED_D + col] = acc[i][j][r];
            }
        }
}

// ---------------------------------------------------------------- sparse scatter-add
// one wave per edge: agg[row] += val * support[col]
__global__ __launch_bounds__(256) void spmm_scatter_kernel(
    const int*   __restrict__ rows,
    const int*   __restrict__ cols,
    const float* __restrict__ vals,
    const float* __restrict__ support,
    float*       __restrict__ agg)
{
    int wave = threadIdx.x >> 5;
    int lane = threadIdx.x & 31;
    long e = (long)blockIdx.x * 8 + wave;
    int r = rows[e];
    int c = cols[e];
    float v = vals[e];
    const float4* src = (const float4*)(support + (long)c * EMBED_D);
    float*        dst = agg + (long)r * EMBED_D;
    #pragma unroll
    for (int i = 0; i < 4; ++i) {
        int idx4 = lane + i * 32;
        float4 m = src[idx4];
        int base = idx4 * 4;
        atomicAdd(dst + base + 0, v * m.x);
        atomicAdd(dst + base + 1, v * m.y);
        atomicAdd(dst + base + 2, v * m.z);
        atomicAdd(dst + base + 3, v * m.w);
    }
}

// ---------------------------------------------------------------- bias + LayerNorm
// one wave per 512-wide row
__global__ __launch_bounds__(256) void bias_ln_kernel(
    const float* __restrict__ agg,
    const float* __restrict__ bias,
    const float* __restrict__ gamma,
    const float* __restrict__ beta,
    float*       __restrict__ out)
{
    int wave = threadIdx.x >> 5;
    int lane = threadIdx.x & 31;
    long row = (long)blockIdx.x * 8 + wave;
    const float4* src = (const float4*)(agg + row * EMBED_D);
    const float4* b4  = (const float4*)bias;
    float4 x[4];
    float s = 0.f, ss = 0.f;
    #pragma unroll
    for (int i = 0; i < 4; ++i) {
        int idx4 = lane + i * 32;
        float4 a = src[idx4];
        float4 bb = b4[idx4];
        a.x += bb.x; a.y += bb.y; a.z += bb.z; a.w += bb.w;
        x[i] = a;
        s  += a.x + a.y + a.z + a.w;
        ss += a.x * a.x + a.y * a.y + a.z * a.z + a.w * a.w;
    }
    #pragma unroll
    for (int off = 16; off > 0; off >>= 1) {
        s  += __shfl_xor(s,  off, 32);
        ss += __shfl_xor(ss, off, 32);
    }
    const float inv = 1.0f / (float)EMBED_D;
    float mu  = s * inv;
    float var = ss * inv - mu * mu;
    float rs  = rsqrtf(var + 1e-5f);
    const float4* g4  = (const float4*)gamma;
    const float4* be4 = (const float4*)beta;
    float4* dst = (float4*)(out + row * EMBED_D);
    #pragma unroll
    for (int i = 0; i < 4; ++i) {
        int idx4 = lane + i * 32;
        float4 g = g4[idx4];
        float4 be = be4[idx4];
        float4 o;
        o.x = (x[i].x - mu) * rs * g.x + be.x;
        o.y = (x[i].y - mu) * rs * g.y + be.y;
        o.z = (x[i].z - mu) * rs * g.z + be.z;
        o.w = (x[i].w - mu) * rs * g.w + be.w;
        dst[idx4] = o;
    }
}

// ---------------------------------------------------------------- launch
extern "C" void kernel_launch(void* const* d_in, const int* in_sizes, int n_in,
                              void* d_out, int out_size, void* d_ws, size_t ws_size,
                              hipStream_t stream) {
    (void)in_sizes; (void)n_in; (void)out_size; (void)ws_size;
    const int*   user_ids  = (const int*)  d_in[0];
    const int*   adj_row   = (const int*)  d_in[1];
    const int*   adj_col   = (const int*)  d_in[2];
    const float* adj_vals  = (const float*)d_in[3];
    const float* embedding = (const float*)d_in[4];
    const float* W         = (const float*)d_in[5];
    const float* bias      = (const float*)d_in[6];
    const float* ln_gamma  = (const float*)d_in[7];
    const float* ln_beta   = (const float*)d_in[8];

    float* out = (float*)d_out;                       // also doubles as `support`
    // workspace layout: [0, 64MB) agg f32 ; then W in bf16 (512 KB)
    float*          agg = (float*)d_ws;
    unsigned short* Wbf = (unsigned short*)((char*)d_ws + (size_t)N_NODES * EMBED_D * sizeof(float));

    // 1. W -> bf16
    convert_w_kernel<<<256, 256, 0, stream>>>(W, Wbf);
    // 2. zero agg (32768*512 floats = 4M float4)
    zero_kernel<<<(N_NODES * EMBED_D / 4) / 256, 256, 0, stream>>>((float4*)agg);
    // 3. gather + GEMM (support -> d_out)
    dim3 ggrid(EMBED_D / BN, N_NODES / BM);           // (8, 256)
    gemm_gather_kernel<<<ggrid, 256, 0, stream>>>(user_ids, embedding, Wbf, out);
    // 4. sparse aggregation: agg += A_sp @ support
    spmm_scatter_kernel<<<NNZ / 8, 256, 0, stream>>>(adj_row, adj_col, adj_vals, out, agg);
    // 5. bias + LayerNorm -> d_out
    bias_ln_kernel<<<N_NODES / 8, 256, 0, stream>>>(agg, bias, ln_gamma, ln_beta, out);
}
